// EquivariantSelfAttentionBlock_38663295598609
// MI455X (gfx1250) — matrix operand order repaired and checked by reading, not compiled, not run on hardware
//
#include <hip/hip_runtime.h>
#include <hip/hip_bf16.h>

typedef unsigned short u16;
typedef unsigned int   u32;

typedef __attribute__((ext_vector_type(16))) __bf16 v16bf;
typedef __attribute__((ext_vector_type(8)))  float  v8f;

union Frag { v16bf v; u16 u[16]; u32 d[8]; uint4 q[2]; };

__device__ __forceinline__ u16 f2bf(float f) {
    u32 x = __builtin_bit_cast(u32, f);
    u32 r = (x + 0x7FFFu + ((x >> 16) & 1u)) >> 16;   // round-to-nearest-even
    return (u16)r;
}
__device__ __forceinline__ float gelu_tanh(float x) {
    const float c = 0.7978845608028654f;               // sqrt(2/pi)
    return 0.5f * x * (1.0f + tanhf(c * (x + 0.044715f * x * x * x)));
}

// ---------------------------------------------------------------- f32 [K][N] -> bf16 [N][K] (transposed)
__global__ void cvt_bf16_T_kernel(const float* __restrict__ src, u16* __restrict__ dst,
                                  int K, int N) {
    int i = blockIdx.x * blockDim.x + threadIdx.x;
    if (i < K * N) {
        int k = i / N, n = i % N;
        dst[(size_t)n * K + k] = f2bf(src[i]);
    }
}

// ---------------------------------------------------------------- LayerNorm (C=256, one block per row)
__global__ __launch_bounds__(256)
void layernorm_kernel(const float* __restrict__ a, const float* __restrict__ sc,
                      const float* __restrict__ bi, u16* __restrict__ out) {
    int row = blockIdx.x;
    int t = threadIdx.x;
    float x = a[(size_t)row * 256 + t];
    float s = x, s2 = x * x;
    #pragma unroll
    for (int off = 16; off > 0; off >>= 1) {
        s  += __shfl_xor(s, off, 32);
        s2 += __shfl_xor(s2, off, 32);
    }
    __shared__ float ps[8], ps2[8];
    int wv = t >> 5;
    if ((t & 31) == 0) { ps[wv] = s; ps2[wv] = s2; }
    __syncthreads();
    float ts = 0.f, ts2 = 0.f;
    #pragma unroll
    for (int i = 0; i < 8; ++i) { ts += ps[i]; ts2 += ps2[i]; }
    float mu  = ts * (1.0f / 256.0f);
    float var = ts2 * (1.0f / 256.0f) - mu * mu;
    float rstd = rsqrtf(var + 1e-6f);
    out[(size_t)row * 256 + t] = f2bf((x - mu) * rstd * sc[t] + bi[t]);
}

// ---------------------------------------------------------------- generic bf16 WMMA GEMM
// C[M,N] = A[M,K] @ W[K,N] with W pre-transposed as WT[N,K] (bf16).
// 4 waves/block; each wave computes a 16x64 strip (A fragment reused 4x).
// Double-buffered fragments: next k-step's loads overlap current WMMAs.
// grid = (N/64, M/64). K must be a multiple of 64 (holds: 256, 2048).
__global__ __launch_bounds__(128)
void gemm_bf16_kernel(const u16* __restrict__ A, const u16* __restrict__ WT,
                      const float* __restrict__ bias, const float* __restrict__ resid,
                      float* __restrict__ outF, u16* __restrict__ outB,
                      int M, int N, int K, int do_gelu) {
    int lane = threadIdx.x & 31;
    int wave = threadIdx.x >> 5;
    int m0 = blockIdx.y * 64 + wave * 16;
    int nbase = blockIdx.x * 64;
    int lhi = (lane >> 4) & 1;
    int l15 = lane & 15;
    v8f acc[4] = {};
    const u16* Arow = A + (size_t)(m0 + l15) * K + lhi * 16;
    const u16* Wrow[4];
    #pragma unroll
    for (int t = 0; t < 4; ++t)
        Wrow[t] = WT + (size_t)(nbase + t * 16 + l15) * K + lhi * 16;

    Frag af0, af1, bf0[4], bf1[4];
    // prologue: k-step 0
    af0.q[0] = *(const uint4*)(Arow);
    af0.q[1] = *(const uint4*)(Arow + 8);
    #pragma unroll
    for (int t = 0; t < 4; ++t) {
        bf0[t].q[0] = *(const uint4*)(Wrow[t]);
        bf0[t].q[1] = *(const uint4*)(Wrow[t] + 8);
    }
    for (int k0 = 0; k0 < K; k0 += 64) {
        // load k-step (k0+32) into buffer 1 while buffer 0 computes
        {
            const u16* ar = Arow + k0 + 32;
            af1.q[0] = *(const uint4*)(ar);
            af1.q[1] = *(const uint4*)(ar + 8);
            #pragma unroll
            for (int t = 0; t < 4; ++t) {
                const u16* wr = Wrow[t] + k0 + 32;
                bf1[t].q[0] = *(const uint4*)(wr);
                bf1[t].q[1] = *(const uint4*)(wr + 8);
            }
            __builtin_prefetch(Arow + k0 + 64, 0, 3);
            __builtin_prefetch(Wrow[0] + k0 + 64, 0, 3);
        }
        #pragma unroll
        for (int t = 0; t < 4; ++t)
            acc[t] = __builtin_amdgcn_wmma_f32_16x16x32_bf16(false, af0.v, false, bf0[t].v,
                                                             (short)0, acc[t], false, false);
        // load k-step (k0+64) into buffer 0 while buffer 1 computes
        if (k0 + 64 < K) {
            const u16* ar = Arow + k0 + 64;
            af0.q[0] = *(const uint4*)(ar);
            af0.q[1] = *(const uint4*)(ar + 8);
            #pragma unroll
            for (int t = 0; t < 4; ++t) {
                const u16* wr = Wrow[t] + k0 + 64;
                bf0[t].q[0] = *(const uint4*)(wr);
                bf0[t].q[1] = *(const uint4*)(wr + 8);
            }
        }
        #pragma unroll
        for (int t = 0; t < 4; ++t)
            acc[t] = __builtin_amdgcn_wmma_f32_16x16x32_bf16(false, af1.v, false, bf1[t].v,
                                                             (short)0, acc[t], false, false);
    }
    #pragma unroll
    for (int t = 0; t < 4; ++t) {
        int n = nbase + t * 16 + l15;
        float bb = bias ? bias[n] : 0.0f;
        #pragma unroll
        for (int r = 0; r < 8; ++r) {
            int m = m0 + (lhi ? 8 : 0) + r;
            float v = acc[t][r] + bb;
            if (resid) v += resid[(size_t)m * N + n];
            if (do_gelu) v = gelu_tanh(v);
            if (outF) outF[(size_t)m * N + n] = v;
            if (outB) outB[(size_t)m * N + n] = f2bf(v);
        }
    }
}

// ---------------------------------------------------------------- pair RFF-MLP bias + window term
// biasT[b,q,k,h] = MLP(sin/cos(2pi * rel @ rffB)) - |rel|^2 / (2 sigma^2)
__global__ __launch_bounds__(256)
void pair_bias_kernel(const float* __restrict__ p, const float* __restrict__ wsize,
                      const float* __restrict__ rffB, const float* __restrict__ Wp1,
                      const float* __restrict__ bp1, const float* __restrict__ Wp2,
                      const float* __restrict__ bp2, float* __restrict__ biasT,
                      int B_, int N_) {
    __shared__ float sW1[32 * 32];
    __shared__ float sW2[32 * 8];
    __shared__ float sb1[32], sb2[8], sB[32];
    int t = threadIdx.x;
    if (t < 32) { sb1[t] = bp1[t]; sB[t] = rffB[t]; }
    if (t < 8)  sb2[t] = bp2[t];
    for (int i = t; i < 1024; i += 256) sW1[i] = Wp1[i];
    sW2[t] = Wp2[t];
    __syncthreads();

    long idx = (long)blockIdx.x * 256 + t;
    int k = (int)(idx % N_);
    int q = (int)((idx / N_) % N_);
    int b = (int)(idx / ((long)N_ * N_));

    float px = p[((size_t)b * N_ + q) * 2 + 0] - p[((size_t)b * N_ + k) * 2 + 0];
    float py = p[((size_t)b * N_ + q) * 2 + 1] - p[((size_t)b * N_ + k) * 2 + 1];
    float sg = wsize[0];
    float wterm = (px * px + py * py) / (2.0f * sg * sg);

    const float TWO_PI = 6.283185307179586f;
    float emb[32];
    #pragma unroll
    for (int f = 0; f < 16; ++f) {
        float ang = TWO_PI * (px * sB[f] + py * sB[16 + f]);
        emb[f]      = sinf(ang);
        emb[16 + f] = cosf(ang);
    }
    float h1[32];
    #pragma unroll
    for (int ph = 0; ph < 32; ++ph) {
        float s = sb1[ph];
        #pragma unroll
        for (int e = 0; e < 32; ++e) s += emb[e] * sW1[e * 32 + ph];
        h1[ph] = gelu_tanh(s);
    }
    size_t base = (((size_t)b * N_ + q) * N_ + k) * 8;
    #pragma unroll
    for (int hh = 0; hh < 8; ++hh) {
        float s = sb2[hh];
        #pragma unroll
        for (int e = 0; e < 32; ++e) s += h1[e] * sW2[e * 8 + hh];
        biasT[base + hh] = s - wterm;
    }
}

// ---------------------------------------------------------------- fused attention
// One block per (b, h, 16-query tile). 8 waves. Logits + bf16 P + transposed V tile in LDS.
#define SROW 1028   // fp32 logits row stride (floats): bank-conflict-free
#define PROW 1032   // bf16 P row stride (u16): 2064B, 16B aligned, bank-shifted
#define VROW 40     // transposed V tile row stride (u16): 80B, 16B aligned
__global__ __launch_bounds__(256)
void attn_kernel(const u16* __restrict__ Qb, const u16* __restrict__ Kb,
                 const u16* __restrict__ Vb, const float* __restrict__ biasT,
                 u16* __restrict__ Ob, int B_, int N_) {
    extern __shared__ char smem[];
    float* S  = (float*)smem;                                    // [16][SROW] fp32 logits
    u16*  Pb  = (u16*)(smem + 16 * SROW * sizeof(float));        // [16][PROW] bf16 probs
    u16*  VtT = (u16*)(smem + 16 * SROW * sizeof(float)
                            + 16 * PROW * sizeof(u16));          // [256][VROW] V^T tile
    const int H_ = 8, DH_ = 256;
    int b = blockIdx.z, h = blockIdx.y, q0 = blockIdx.x * 16;
    int lane = threadIdx.x & 31, wave = threadIdx.x >> 5;
    int lhi = (lane >> 4) & 1, l15 = lane & 15;

    // ---- Phase 1: S = Q K^T * scale + bias ----
    Frag qf[8];
    const u16* Qrow = Qb + ((size_t)(b * N_ + q0 + l15) * H_ + h) * DH_;
    #pragma unroll
    for (int s = 0; s < 8; ++s) {
        const u16* qp = Qrow + s * 32 + lhi * 16;
        qf[s].q[0] = *(const uint4*)(qp);
        qf[s].q[1] = *(const uint4*)(qp + 8);
    }
    for (int t = wave; t < 64; t += 8) {            // 64 key tiles, 8 per wave
        int k0 = t * 16;
        const u16* Krow = Kb + ((size_t)(b * N_ + k0 + l15) * H_ + h) * DH_ + lhi * 16;
        v8f acc = {};
        Frag kf0, kf1;                              // double-buffered K fragments
        kf0.q[0] = *(const uint4*)(Krow);
        kf0.q[1] = *(const uint4*)(Krow + 8);
        #pragma unroll
        for (int s = 0; s < 8; s += 2) {
            const u16* kp1 = Krow + (s + 1) * 32;
            kf1.q[0] = *(const uint4*)(kp1);
            kf1.q[1] = *(const uint4*)(kp1 + 8);
            acc = __builtin_amdgcn_wmma_f32_16x16x32_bf16(false, qf[s].v, false, kf0.v,
                                                          (short)0, acc, false, false);
            if (s + 2 < 8) {
                const u16* kp2 = Krow + (s + 2) * 32;
                kf0.q[0] = *(const uint4*)(kp2);
                kf0.q[1] = *(const uint4*)(kp2 + 8);
            }
            acc = __builtin_amdgcn_wmma_f32_16x16x32_bf16(false, qf[s + 1].v, false, kf1.v,
                                                          (short)0, acc, false, false);
        }
        const float scale = 0.0625f;                // 1/sqrt(256)
        #pragma unroll
        for (int r = 0; r < 8; ++r) {
            int m = (lhi ? 8 : 0) + r;
            int kk = k0 + l15;
            float v = acc[r] * scale +
                      biasT[(((size_t)(b * N_ + q0 + m)) * N_ + kk) * 8 + h];
            S[m * SROW + kk] = v;
        }
    }
    __syncthreads();

    // ---- Phase 2: softmax over keys, 2 rows per wave; write normalized bf16 P ----
    for (int rr = 0; rr < 2; ++rr) {
        int row = wave * 2 + rr;
        float mx = -1e30f;
        for (int c = lane; c < N_; c += 32) mx = fmaxf(mx, S[row * SROW + c]);
        #pragma unroll
        for (int off = 16; off > 0; off >>= 1) mx = fmaxf(mx, __shfl_xor(mx, off, 32));
        float sum = 0.f;
        for (int c = lane; c < N_; c += 32) {
            float e = expf(S[row * SROW + c] - mx);
            S[row * SROW + c] = e;
            sum += e;
        }
        #pragma unroll
        for (int off = 16; off > 0; off >>= 1) sum += __shfl_xor(sum, off, 32);
        float inv = 1.0f / sum;
        for (int c = lane; c < N_; c += 32)
            Pb[row * PROW + c] = f2bf(S[row * SROW + c] * inv);
    }
    __syncthreads();

    // ---- Phase 3: O = P V. Wave w owns dh range [w*32, w*32+32). ----
    v8f acc0 = {}, acc1 = {};
    int n0a = wave * 32;
    for (int kb = 0; kb < 32; ++kb) {
        int kk0 = kb * 32;
        __syncthreads();
        // stage V tile transposed: VtT[dh][key], global reads coalesced along dh
        for (int i = threadIdx.x; i < 4096; i += 256) {
            int key = i >> 7;
            int dhp = (i & 127) * 2;
            u32 val = *(const u32*)(Vb + ((size_t)(b * N_ + kk0 + key) * H_ + h) * DH_ + dhp);
            VtT[(dhp)     * VROW + key] = (u16)(val & 0xffffu);
            VtT[(dhp + 1) * VROW + key] = (u16)(val >> 16);
        }
        __syncthreads();
        Frag pf, vf0, vf1;
        // A: P rows = queries, K = key idx; two contiguous 8-elem runs -> 2x ds b128
        const u16* prow = Pb + l15 * PROW + kk0 + lhi * 16;
        pf.q[0] = *(const uint4*)(prow);
        pf.q[1] = *(const uint4*)(prow + 8);
        // B: V^T rows = dh, 16 contiguous key elems per lane -> 2x ds b128
        const u16* v0row = VtT + (n0a + l15) * VROW + lhi * 16;
        vf0.q[0] = *(const uint4*)(v0row);
        vf0.q[1] = *(const uint4*)(v0row + 8);
        const u16* v1row = VtT + (n0a + 16 + l15) * VROW + lhi * 16;
        vf1.q[0] = *(const uint4*)(v1row);
        vf1.q[1] = *(const uint4*)(v1row + 8);
        acc0 = __builtin_amdgcn_wmma_f32_16x16x32_bf16(false, pf.v, false, vf0.v,
                                                       (short)0, acc0, false, false);
        acc1 = __builtin_amdgcn_wmma_f32_16x16x32_bf16(false, pf.v, false, vf1.v,
                                                       (short)0, acc1, false, false);
    }
    #pragma unroll
    for (int r = 0; r < 8; ++r) {
        int m = (lhi ? 8 : 0) + r;
        size_t rowo = (size_t)(b * N_ + q0 + m) * 2048 + (size_t)h * DH_;
        Ob[rowo + n0a + l15]      = f2bf(acc0[r]);
        Ob[rowo + n0a + 16 + l15] = f2bf(acc1[r]);
    }
}

// ================================================================ launcher
extern "C" void kernel_launch(void* const* d_in, const int* in_sizes, int n_in,
                              void* d_out, int out_size, void* d_ws, size_t ws_size,
                              hipStream_t stream) {
    (void)in_sizes; (void)n_in; (void)out_size; (void)ws_size;
    const int B = 2, N = 1024, C = 256, H = 8, DH = 256;
    const int M = B * N;       // 2048 rows
    const int HD = H * DH;     // 2048

    const float* p   = (const float*)d_in[0];
    const float* a   = (const float*)d_in[1];
    const float* wsz = (const float*)d_in[2];
    const float* lns = (const float*)d_in[3];
    const float* lnb = (const float*)d_in[4];
    const float* Wq  = (const float*)d_in[5];
    const float* bq  = (const float*)d_in[6];
    const float* Wk  = (const float*)d_in[7];
    const float* bk  = (const float*)d_in[8];
    const float* Wv  = (const float*)d_in[9];
    const float* bv  = (const float*)d_in[10];
    const float* rffB= (const float*)d_in[11];
    const float* Wp1 = (const float*)d_in[12];
    const float* bp1 = (const float*)d_in[13];
    const float* Wp2 = (const float*)d_in[14];
    const float* bp2 = (const float*)d_in[15];
    const float* Wo  = (const float*)d_in[16];
    const float* bo  = (const float*)d_in[17];
    const float* Wf1 = (const float*)d_in[18];
    const float* bf1 = (const float*)d_in[19];
    const float* Wf2 = (const float*)d_in[20];
    const float* bf2 = (const float*)d_in[21];

    char* w = (char*)d_ws;
    auto carve = [&](size_t bytes) { char* r = w; w += (bytes + 255) & ~(size_t)255; return r; };
    u16* xnb   = (u16*)carve((size_t)M * C * 2);
    u16* WqT   = (u16*)carve((size_t)C * HD * 2);   // [HD][C]
    u16* WkT   = (u16*)carve((size_t)C * HD * 2);
    u16* WvT   = (u16*)carve((size_t)C * HD * 2);
    u16* WoT   = (u16*)carve((size_t)HD * C * 2);   // [C][HD]
    u16* Wf1T  = (u16*)carve((size_t)C * C * 2);
    u16* Wf2T  = (u16*)carve((size_t)C * C * 2);
    u16* Qb    = (u16*)carve((size_t)M * HD * 2);
    u16* Kb    = (u16*)carve((size_t)M * HD * 2);
    u16* Vb    = (u16*)carve((size_t)M * HD * 2);
    u16* Ob    = (u16*)carve((size_t)M * HD * 2);
    u16* aresb = (u16*)carve((size_t)M * C * 2);
    u16* h1b   = (u16*)carve((size_t)M * C * 2);
    float* biasT = (float*)carve((size_t)B * N * N * H * 4);   // 64 MB

    // transposed weight conversions to bf16
    cvt_bf16_T_kernel<<<(C * HD + 255) / 256, 256, 0, stream>>>(Wq, WqT, C, HD);
    cvt_bf16_T_kernel<<<(C * HD + 255) / 256, 256, 0, stream>>>(Wk, WkT, C, HD);
    cvt_bf16_T_kernel<<<(C * HD + 255) / 256, 256, 0, stream>>>(Wv, WvT, C, HD);
    cvt_bf16_T_kernel<<<(HD * C + 255) / 256, 256, 0, stream>>>(Wo, WoT, HD, C);
    cvt_bf16_T_kernel<<<(C * C + 255) / 256, 256, 0, stream>>>(Wf1, Wf1T, C, C);
    cvt_bf16_T_kernel<<<(C * C + 255) / 256, 256, 0, stream>>>(Wf2, Wf2T, C, C);

    // LayerNorm -> bf16 activations
    layernorm_kernel<<<M, 256, 0, stream>>>(a, lns, lnb, xnb);

    // QKV projections
    gemm_bf16_kernel<<<dim3(HD / 64, M / 64), 128, 0, stream>>>(xnb, WqT, bq, nullptr, nullptr, Qb, M, HD, C, 0);
    gemm_bf16_kernel<<<dim3(HD / 64, M / 64), 128, 0, stream>>>(xnb, WkT, bk, nullptr, nullptr, Kb, M, HD, C, 0);
    gemm_bf16_kernel<<<dim3(HD / 64, M / 64), 128, 0, stream>>>(xnb, WvT, bv, nullptr, nullptr, Vb, M, HD, C, 0);

    // pair bias (+ gaussian window folded in)
    pair_bias_kernel<<<(B * N * N) / 256, 256, 0, stream>>>(p, wsz, rffB, Wp1, bp1, Wp2, bp2, biasT, B, N);

    // fused attention
    size_t smem = 16 * SROW * sizeof(float) + 16 * PROW * sizeof(u16) + 256 * VROW * sizeof(u16);
    attn_kernel<<<dim3(N / 16, H, B), 256, smem, stream>>>(Qb, Kb, Vb, biasT, Ob, B, N);

    // output projection + residual (a), then FFN
    gemm_bf16_kernel<<<dim3(C / 64, M / 64), 128, 0, stream>>>(Ob, WoT, bo, a, nullptr, aresb, M, C, HD, 0);
    gemm_bf16_kernel<<<dim3(C / 64, M / 64), 128, 0, stream>>>(aresb, Wf1T, bf1, nullptr, nullptr, h1b, M, C, C, 1);
    gemm_bf16_kernel<<<dim3(C / 64, M / 64), 128, 0, stream>>>(h1b, Wf2T, bf2, nullptr, (float*)d_out, nullptr, M, C, C, 0);
}
